// CfdGinoMeshToGridOld_49744311222692
// MI455X (gfx1250) — matrix operand order repaired and checked by reading, not compiled, not run on hardware
//
#include <hip/hip_runtime.h>
#include <hip/hip_bf16.h>

// ---------------- problem constants ----------------
#define IN_DIM   16
#define HID      384
#define POS_DIM  192        // HID/2
#define NMESH    50000
#define MPAD     50048      // 391 * 128
#define GRID_N   32768
#define NEDGE    131072
#define EDIM     768        // 2*HID

typedef __attribute__((ext_vector_type(16))) __bf16 v16bf;
typedef __attribute__((ext_vector_type(8)))  float  v8f;

union Frag16 { v16bf v; uint4 q[2]; };

__device__ __forceinline__ float gelu_exact(float x) {
    return 0.5f * x * (1.0f + erff(x * 0.70710678118654752f));
}

// ---------------------------------------------------------------
// Pack fp32 weight (K x N row-major) into WMMA-B fragment-major
// bf16: out[((nt*numKS + ks)*32 + lane)*16 + elem], where
//   n = nt*16 + (lane&15), k = ks*32 + (lane>>4)*8 + (elem&7) + (elem>>3)*16
// matching the 16-bit B-matrix (K x 16) VGPR layout.
// ---------------------------------------------------------------
__global__ __launch_bounds__(256)
void pack_weight_kernel(const float* __restrict__ W, __bf16* __restrict__ out,
                        int K, int N)
{
    long long i = (long long)blockIdx.x * blockDim.x + threadIdx.x;
    long long total = (long long)K * N;
    if (i >= total) return;
    int elem = (int)(i & 15);
    int lane = (int)((i >> 4) & 31);
    long long t = i >> 9;            // nt*numKS + ks
    int numKS = K >> 5;
    int nt = (int)(t / numKS);
    int ks = (int)(t % numKS);
    int n  = nt * 16 + (lane & 15);
    int k  = ks * 32 + (lane >> 4) * 8 + (elem & 7) + (elem >> 3) * 16;
    out[i] = (__bf16)W[(long long)k * N + n];
}

// ---------------------------------------------------------------
// Generic WMMA GEMM: C[M x N] = act(A[M x K] @ W[K x N] + bias)
//   A: bf16 row-major (lda == K), M multiple of 128, K multiple of 32,
//   W: fragment-packed bf16, N multiple of 64.
// Block = 128 threads (4 waves). Wave w owns TWO M-tiles
// (blockIdx.x*8 + 2w, +1) and the 4 N-tiles of group blockIdx.y, so
// each K-step is 12 b128 loads feeding 8 WMMAs (B reused 2x from regs).
// All 4 B fragments live in distinct registers so the loads of a whole
// K-step can be in flight while the WMMA burst executes.
// STORE==0: bf16 store (ldc stride, col offset).
// STORE==1: f32 atomic scatter-add into Csum[edges[row*2]*HID + n].
// ---------------------------------------------------------------
template<int ACT, int STORE>
__global__ __launch_bounds__(128)
void wmma_gemm_bf16(const __bf16* __restrict__ A, int K,
                    const __bf16* __restrict__ Bp,
                    const float* __restrict__ bias,
                    __bf16* __restrict__ Cbf, int ldc, int coloff,
                    float* __restrict__ Csum, const int* __restrict__ edges)
{
    const int lane = threadIdx.x & 31;
    const int wave = threadIdx.x >> 5;
    const int half = lane >> 4;
    const int l16  = lane & 15;
    const int mt0  = blockIdx.x * 8 + wave * 2;     // first of two M-tiles
    const int numKS = K >> 5;

    const __bf16* __restrict__ arow0 = A + (long long)(mt0 * 16 + l16) * K;
    const __bf16* __restrict__ arow1 = arow0 + (long long)16 * K;

    v8f acc[2][4];
    #pragma unroll
    for (int j = 0; j < 4; ++j) {
        const int n = (blockIdx.y * 4 + j) * 16 + l16;
        const float b0 = bias[n];
        #pragma unroll
        for (int r = 0; r < 8; ++r) { acc[0][j][r] = b0; acc[1][j][r] = b0; }
    }

    for (int ks = 0; ks < numKS; ++ks) {
        // A fragments: lane holds K = ks*32 + half*8 .. +7  and  +16 .. +23
        const int kb = ks * 32 + half * 8;
        Frag16 a0, a1;
        a0.q[0] = *(const uint4*)(arow0 + kb);
        a0.q[1] = *(const uint4*)(arow0 + kb + 16);
        a1.q[0] = *(const uint4*)(arow1 + kb);
        a1.q[1] = *(const uint4*)(arow1 + kb + 16);

        // All four B fragments up front, in distinct registers.
        Frag16 b[4];
        #pragma unroll
        for (int j = 0; j < 4; ++j) {
            const long long nt = blockIdx.y * 4 + j;
            const __bf16* __restrict__ bp =
                Bp + (((nt * numKS) + ks) * 32 + lane) * 16;
            b[j].q[0] = *(const uint4*)(bp);
            b[j].q[1] = *(const uint4*)(bp + 8);
        }

        #pragma unroll
        for (int j = 0; j < 4; ++j) {
            acc[0][j] = __builtin_amdgcn_wmma_f32_16x16x32_bf16(
                false, a0.v, false, b[j].v, (short)0, acc[0][j], false, false);
            acc[1][j] = __builtin_amdgcn_wmma_f32_16x16x32_bf16(
                false, a1.v, false, b[j].v, (short)0, acc[1][j], false, false);
        }
    }

    if (STORE == 0) {
        #pragma unroll
        for (int t = 0; t < 2; ++t) {
            #pragma unroll
            for (int j = 0; j < 4; ++j) {
                const int n = (blockIdx.y * 4 + j) * 16 + l16 + coloff;
                #pragma unroll
                for (int r = 0; r < 8; ++r) {
                    const int row = (mt0 + t) * 16 + half * 8 + r;
                    float v = acc[t][j][r];
                    if (ACT) v = gelu_exact(v);
                    Cbf[(long long)row * ldc + n] = (__bf16)v;
                }
            }
        }
    } else {
        #pragma unroll
        for (int t = 0; t < 2; ++t) {
            #pragma unroll
            for (int j = 0; j < 4; ++j) {
                const int n = (blockIdx.y * 4 + j) * 16 + l16;
                #pragma unroll
                for (int r = 0; r < 8; ++r) {
                    const int row = (mt0 + t) * 16 + half * 8 + r;
                    const int g = edges[(long long)row * 2];   // grid index
                    atomicAdd(&Csum[(long long)g * HID + n], acc[t][j][r]);
                }
            }
        }
    }
}

// ---------------------------------------------------------------
// Mesh layer 1: K=16 too small for WMMA; trivial FLOPs, do directly.
// h1[m][n] = gelu(x[m] . w1[:,n] + b1[n]) as bf16.
// ---------------------------------------------------------------
__global__ __launch_bounds__(256)
void mesh_layer1_kernel(const float* __restrict__ x,
                        const float* __restrict__ w,
                        const float* __restrict__ b,
                        __bf16* __restrict__ h1)
{
    long long i = (long long)blockIdx.x * blockDim.x + threadIdx.x;
    if (i >= (long long)NMESH * HID) return;
    int m = (int)(i / HID), n = (int)(i % HID);
    float acc = b[n];
    #pragma unroll
    for (int k = 0; k < IN_DIM; ++k)
        acc += x[m * IN_DIM + k] * w[k * HID + n];
    h1[(long long)m * HID + n] = (__bf16)gelu_exact(acc);
}

// ---------------------------------------------------------------
// Sin/cos positional embedding (192 values per item), bf16 output.
// emb[d*64 + j] = sin(c_d * omega_j) (j<32)  / cos (j>=32)
// omega_j = 10000^(-(2*(j%32))/64)
// ---------------------------------------------------------------
__global__ __launch_bounds__(256)
void embed_kernel(const float* __restrict__ pos, __bf16* __restrict__ out,
                  int n_items, int row_stride, int col_off)
{
    long long i = (long long)blockIdx.x * blockDim.x + threadIdx.x;
    if (i >= (long long)n_items * POS_DIM) return;
    int m = (int)(i / POS_DIM), j = (int)(i % POS_DIM);
    int d = j >> 6, jj = j & 63;
    float c = pos[m * 3 + d];
    float omega = powf(10000.0f, -(2.0f * (float)(jj & 31)) / 64.0f);
    float v = c * omega;
    float r = (jj < 32) ? sinf(v) : cosf(v);
    out[(long long)m * row_stride + col_off + j] = (__bf16)r;
}

// ---------------------------------------------------------------
// Gather edge features: e[ed] = [mesh_feat[mesh_idx] (576) | grid_emb[gi] (192)]
// 4 bf16 (8 bytes) per thread; 576 % 4 == 0 so chunks never straddle.
// ---------------------------------------------------------------
__global__ __launch_bounds__(256)
void gather_edges_kernel(const int* __restrict__ edges,
                         const __bf16* __restrict__ mesh_feat,
                         const __bf16* __restrict__ grid_emb,
                         __bf16* __restrict__ e)
{
    long long i = (long long)blockIdx.x * blockDim.x + threadIdx.x;
    if (i >= (long long)NEDGE * 192) return;
    int c = (int)(i % 192);
    long long ed = i / 192;
    const __bf16* src;
    if (c < 144) {
        int mi = edges[ed * 2 + 1];
        src = mesh_feat + (long long)mi * (HID + POS_DIM) + c * 4;
    } else {
        int gi = edges[ed * 2 + 0];
        src = grid_emb + (long long)gi * POS_DIM + (c - 144) * 4;
    }
    *(uint2*)(e + ed * EDIM + c * 4) = *(const uint2*)src;
}

__global__ __launch_bounds__(256)
void zero_kernel(float* __restrict__ p, long long n)
{
    long long i = (long long)blockIdx.x * blockDim.x + threadIdx.x;
    if (i < n) p[i] = 0.0f;
}

__global__ __launch_bounds__(256)
void counts_kernel(const int* __restrict__ edges, float* __restrict__ cnt)
{
    int e = blockIdx.x * blockDim.x + threadIdx.x;
    if (e >= NEDGE) return;
    atomicAdd(&cnt[edges[(long long)e * 2]], 1.0f);
}

__global__ __launch_bounds__(256)
void finalize_kernel(const float* __restrict__ sums,
                     const float* __restrict__ cnt,
                     float* __restrict__ out)
{
    long long i = (long long)blockIdx.x * blockDim.x + threadIdx.x;
    if (i >= (long long)GRID_N * HID) return;
    out[i] = sums[i] / fmaxf(cnt[i / HID], 1.0f);
}

// ---------------------------------------------------------------
extern "C" void kernel_launch(void* const* d_in, const int* in_sizes, int n_in,
                              void* d_out, int out_size, void* d_ws, size_t ws_size,
                              hipStream_t stream)
{
    const float* x        = (const float*)d_in[0];
    const float* mesh_pos = (const float*)d_in[1];
    const float* grid_pos = (const float*)d_in[2];
    const int*   edges    = (const int*)d_in[3];
    const float* w_in1 = (const float*)d_in[4];
    const float* b_in1 = (const float*)d_in[5];
    const float* w_in2 = (const float*)d_in[6];
    const float* b_in2 = (const float*)d_in[7];
    const float* w_in3 = (const float*)d_in[8];
    const float* b_in3 = (const float*)d_in[9];
    const float* w_m1  = (const float*)d_in[10];
    const float* b_m1  = (const float*)d_in[11];
    const float* w_m2  = (const float*)d_in[12];
    const float* b_m2  = (const float*)d_in[13];
    const float* w_m3  = (const float*)d_in[14];
    const float* b_m3  = (const float*)d_in[15];
    float* out = (float*)d_out;
    char*  ws  = (char*)d_ws;

    // ---- workspace layout (bytes) ----
    // Region A (reused): first holds h1|h2 (mesh), later holds e0 (edges)
    const size_t sz_e   = (size_t)NEDGE * EDIM * 2;          // 201.3 MB
    const size_t sz_h   = (size_t)MPAD  * HID  * 2;          // 38.4 MB
    __bf16* h1  = (__bf16*)(ws + 0);
    __bf16* h2  = (__bf16*)(ws + ((sz_h + 255) & ~(size_t)255));
    __bf16* e0  = (__bf16*)(ws + 0);                         // reuse after mesh
    size_t off = (sz_e + 255) & ~(size_t)255;
    __bf16* e1  = (__bf16*)(ws + off); off += (sz_e + 255) & ~(size_t)255;
    __bf16* mesh_feat = (__bf16*)(ws + off);
    off += (((size_t)MPAD * (HID + POS_DIM) * 2) + 255) & ~(size_t)255;
    __bf16* grid_emb  = (__bf16*)(ws + off);
    off += (((size_t)GRID_N * POS_DIM * 2) + 255) & ~(size_t)255;
    float* sums = (float*)(ws + off);
    off += (((size_t)GRID_N * HID * 4) + 255) & ~(size_t)255;
    float* cnt  = (float*)(ws + off);
    off += (((size_t)GRID_N * 4) + 255) & ~(size_t)255;
    __bf16* wp_in2 = (__bf16*)(ws + off); off += ((size_t)HID * HID * 2 + 255) & ~(size_t)255;
    __bf16* wp_in3 = (__bf16*)(ws + off); off += ((size_t)HID * HID * 2 + 255) & ~(size_t)255;
    __bf16* wp_m1  = (__bf16*)(ws + off); off += ((size_t)EDIM * EDIM * 2 + 255) & ~(size_t)255;
    __bf16* wp_m2  = (__bf16*)(ws + off); off += ((size_t)EDIM * EDIM * 2 + 255) & ~(size_t)255;
    __bf16* wp_m3  = (__bf16*)(ws + off);

    const int TPB = 256;
    auto blks = [](long long n, int t) { return (unsigned)((n + t - 1) / t); };

    // ---- pack weights into WMMA-B fragment layout (bf16) ----
    pack_weight_kernel<<<blks((long long)HID * HID, TPB), TPB, 0, stream>>>(w_in2, wp_in2, HID, HID);
    pack_weight_kernel<<<blks((long long)HID * HID, TPB), TPB, 0, stream>>>(w_in3, wp_in3, HID, HID);
    pack_weight_kernel<<<blks((long long)EDIM * EDIM, TPB), TPB, 0, stream>>>(w_m1, wp_m1, EDIM, EDIM);
    pack_weight_kernel<<<blks((long long)EDIM * EDIM, TPB), TPB, 0, stream>>>(w_m2, wp_m2, EDIM, EDIM);
    pack_weight_kernel<<<blks((long long)EDIM * HID, TPB), TPB, 0, stream>>>(w_m3, wp_m3, EDIM, HID);

    // ---- zero the segment-sum buffers (ws is poisoned) ----
    zero_kernel<<<blks((long long)GRID_N * HID, TPB), TPB, 0, stream>>>(sums, (long long)GRID_N * HID);
    zero_kernel<<<blks(GRID_N, TPB), TPB, 0, stream>>>(cnt, GRID_N);

    // ---- mesh MLP ----
    mesh_layer1_kernel<<<blks((long long)NMESH * HID, TPB), TPB, 0, stream>>>(x, w_in1, b_in1, h1);

    dim3 gm(MPAD / 128, HID / 64);
    wmma_gemm_bf16<1, 0><<<gm, 128, 0, stream>>>(h1, HID, wp_in2, b_in2,
                                                 h2, HID, 0, nullptr, nullptr);
    wmma_gemm_bf16<0, 0><<<gm, 128, 0, stream>>>(h2, HID, wp_in3, b_in3,
                                                 mesh_feat, HID + POS_DIM, 0, nullptr, nullptr);

    // ---- positional embeddings ----
    embed_kernel<<<blks((long long)NMESH * POS_DIM, TPB), TPB, 0, stream>>>(
        mesh_pos, mesh_feat, NMESH, HID + POS_DIM, HID);
    embed_kernel<<<blks((long long)GRID_N * POS_DIM, TPB), TPB, 0, stream>>>(
        grid_pos, grid_emb, GRID_N, POS_DIM, 0);

    // ---- edge feature gather (overwrites h1/h2 region; mesh stage done) ----
    gather_edges_kernel<<<blks((long long)NEDGE * 192, TPB), TPB, 0, stream>>>(
        edges, mesh_feat, grid_emb, e0);

    // ---- edge MLP (the 0.39 TFLOP part) ----
    dim3 ge1(NEDGE / 128, EDIM / 64);
    wmma_gemm_bf16<1, 0><<<ge1, 128, 0, stream>>>(e0, EDIM, wp_m1, b_m1,
                                                  e1, EDIM, 0, nullptr, nullptr);
    wmma_gemm_bf16<1, 0><<<ge1, 128, 0, stream>>>(e1, EDIM, wp_m2, b_m2,
                                                  e0, EDIM, 0, nullptr, nullptr);
    dim3 ge3(NEDGE / 128, HID / 64);
    wmma_gemm_bf16<0, 1><<<ge3, 128, 0, stream>>>(e0, EDIM, wp_m3, b_m3,
                                                  nullptr, 0, 0, sums, edges);

    // ---- segment mean ----
    counts_kernel<<<blks(NEDGE, TPB), TPB, 0, stream>>>(edges, cnt);
    finalize_kernel<<<blks((long long)GRID_N * HID, TPB), TPB, 0, stream>>>(sums, cnt, out);
}
